// SudokuGNN_23545010717334
// MI455X (gfx1250) — matrix operand order repaired and checked by reading, not compiled, not run on hardware
//
#include <hip/hip_runtime.h>

typedef __attribute__((ext_vector_type(16))) _Float16 v16h;
typedef __attribute__((ext_vector_type(8)))  float    v8f;

#define NPUZ   2048
#define EPER   1620
#define NNODES (NPUZ * 81)      // 165888, divisible by 16
#define NTILES (NNODES / 16)    // 10368, divisible by 8

// ---------------------------------------------------------------- degree / norm
__global__ __launch_bounds__(256) void k_init_deg(float* __restrict__ deg) {
    int i = blockIdx.x * 256 + threadIdx.x;
    if (i < NNODES) deg[i] = 1.0f;            // self-loop contributes 1
}

__global__ __launch_bounds__(256) void k_count_deg(const int* __restrict__ ei,
                                                   float* __restrict__ deg) {
    int j = blockIdx.x * 256 + threadIdx.x;
    int b = blockIdx.y;
    if (j < EPER) {
        int dst = ei[(size_t)b * 2 * EPER + EPER + j];
        atomicAdd(&deg[dst], 1.0f);
    }
}

__global__ __launch_bounds__(256) void k_rsqrt_inplace(float* __restrict__ d) {
    int i = blockIdx.x * 256 + threadIdx.x;
    if (i < NNODES) d[i] = rsqrtf(d[i]);      // deg >= 1 always
}

// x: [N,10] f32  ->  h: [N,16] f16 zero-padded
__global__ __launch_bounds__(256) void k_cvt_x(const float* __restrict__ x,
                                               _Float16* __restrict__ h) {
    int i = blockIdx.x * 256 + threadIdx.x;
    if (i < NNODES * 16) {
        int n = i >> 4, c = i & 15;
        h[i] = (c < 10) ? (_Float16)x[n * 10 + c] : (_Float16)0.0f;
    }
}

// ---------------------------------------------------------------- WMMA GEMM
// out[M=node, N=col] = h[M, K] * W[K, N] (+bias) for a 16-node tile per wave.
// A/B packing per CDNA5 ISA 7.12.2 16-bit layouts; C/D: M = r + 8*(lane/16), N = lane%16.
template <int CIN, int KREAL, int COUT, bool OUTH, bool RELU, bool BIAS>
__global__ __launch_bounds__(256) void k_gemm(const _Float16* __restrict__ h,
                                              const float* __restrict__ W,
                                              const float* __restrict__ bias,
                                              float* __restrict__ outF,
                                              _Float16* __restrict__ outH,
                                              int outStride) {
    constexpr int KT = (CIN + 31) / 32;
    constexpr int NT = (COUT + 15) / 16;

    int lane = threadIdx.x & 31;
    int wave = threadIdx.x >> 5;
    int tile = blockIdx.x * 8 + wave;
    if (tile >= NTILES) return;               // uniform per wave: EXEC stays all-1s
    int node0 = tile * 16;
    int lh = lane >> 4;                       // lane half
    int lm = lane & 15;

    // ---- B fragments (weights, tiny: L1/L2 resident)
    v16h bf[KT][NT];
#pragma unroll
    for (int kt = 0; kt < KT; ++kt)
#pragma unroll
        for (int nt = 0; nt < NT; ++nt)
#pragma unroll
            for (int e = 0; e < 16; ++e) {
                int v = e >> 1, hl = e & 1;
                int k = kt * 32 + ((v < 4) ? (2 * v + 8 * lh) : (16 + 2 * (v - 4) + 8 * lh)) + hl;   // same K order as A: lanes16-31 -> K+8
                int n = nt * 16 + lm;
                bf[kt][nt][e] = (k < KREAL && n < COUT)
                                    ? (_Float16)W[k * COUT + n]
                                    : (_Float16)0.0f;
            }

    // ---- A fragments (activations, f16)
    v16h af[KT];
    int row = node0 + lm;
#pragma unroll
    for (int kt = 0; kt < KT; ++kt)
#pragma unroll
        for (int e = 0; e < 16; ++e) {
            int v = e >> 1, hl = e & 1;
            int kb = (v < 4) ? (2 * v + 8 * lh) : (16 + 2 * (v - 4) + 8 * lh);
            int k = kt * 32 + kb + hl;
            af[kt][e] = (k < CIN) ? h[(size_t)row * CIN + k] : (_Float16)0.0f;
        }

    // ---- C init (bias broadcast across rows; col = lane%16)
    v8f cf[NT];
#pragma unroll
    for (int nt = 0; nt < NT; ++nt) {
        int col = nt * 16 + lm;
        float cv = 0.0f;
        if (BIAS && col < COUT) cv = bias[col];
#pragma unroll
        for (int r = 0; r < 8; ++r) cf[nt][r] = cv;
    }

    // ---- MACs
#pragma unroll
    for (int kt = 0; kt < KT; ++kt)
#pragma unroll
        for (int nt = 0; nt < NT; ++nt)
            cf[nt] = __builtin_amdgcn_wmma_f32_16x16x32_f16(
                false, af[kt], false, bf[kt][nt], (short)0, cf[nt], false, false);

    // ---- store
#pragma unroll
    for (int nt = 0; nt < NT; ++nt) {
        int col = nt * 16 + lm;
#pragma unroll
        for (int r = 0; r < 8; ++r) {
            int m = node0 + r + 8 * lh;
            if (col < COUT) {
                float v = cf[nt][r];
                if (RELU) v = fmaxf(v, 0.0f);
                if constexpr (OUTH)
                    outH[(size_t)m * outStride + col] = (_Float16)v;
                else
                    outF[(size_t)m * outStride + col] = v;
            }
        }
    }
}

// ---------------------------------------------------------------- edge aggregation
// acc[dst, c] += dinv[src]*dinv[dst] * hw[src, c]  — thread per (edge, feature)
template <int LOG2C>
__global__ __launch_bounds__(256) void k_aggregate(const int* __restrict__ ei,
                                                   const float* __restrict__ dinv,
                                                   const float* __restrict__ hw,
                                                   float* __restrict__ acc) {
    int idx = blockIdx.x * 256 + threadIdx.x;
    int b = blockIdx.y;
    int j = idx >> LOG2C;
    int c = idx & ((1 << LOG2C) - 1);
    if (j >= EPER) return;
    const int* e = ei + (size_t)b * 2 * EPER;
    int s = e[j];
    int d = e[EPER + j];
    float nrm = dinv[s] * dinv[d];
    atomicAdd(&acc[((size_t)d << LOG2C) + c],
              nrm * hw[((size_t)s << LOG2C) + c]);
}

// h_out = relu(acc + dinv^2 * hw + bias), stored f16 for the next WMMA stage
template <int LOG2C>
__global__ __launch_bounds__(256) void k_finalize(const float* __restrict__ acc,
                                                  const float* __restrict__ hw,
                                                  const float* __restrict__ dinv,
                                                  const float* __restrict__ bias,
                                                  _Float16* __restrict__ hout) {
    int i = blockIdx.x * 256 + threadIdx.x;
    if (i >= (NNODES << LOG2C)) return;
    int n = i >> LOG2C;
    int c = i & ((1 << LOG2C) - 1);
    float di = dinv[n];
    float v = acc[i] + di * di * hw[i] + bias[c];
    hout[i] = (_Float16)fmaxf(v, 0.0f);
}

// ---------------------------------------------------------------- launch
extern "C" void kernel_launch(void* const* d_in, const int* in_sizes, int n_in,
                              void* d_out, int out_size, void* d_ws, size_t ws_size,
                              hipStream_t stream) {
    const float* x  = (const float*)d_in[0];
    const int*   ei = (const int*)d_in[1];
    const float* W1 = (const float*)d_in[2];
    const float* b1 = (const float*)d_in[3];
    const float* W2 = (const float*)d_in[4];
    const float* b2 = (const float*)d_in[5];
    const float* W3 = (const float*)d_in[6];
    const float* b3 = (const float*)d_in[7];
    const float* W4 = (const float*)d_in[8];
    const float* b4 = (const float*)d_in[9];
    const float* W5 = (const float*)d_in[10];
    const float* b5 = (const float*)d_in[11];
    const float* fW1 = (const float*)d_in[12];
    const float* fb1 = (const float*)d_in[13];
    const float* fW2 = (const float*)d_in[14];
    const float* fb2 = (const float*)d_in[15];
    float* out = (float*)d_out;

    // workspace layout
    char* ws = (char*)d_ws;
    size_t off = 0;
    auto alloc = [&](size_t bytes) {
        void* p = ws + off;
        off += (bytes + 255) & ~(size_t)255;
        return p;
    };
    float*    dinv = (float*)alloc((size_t)NNODES * 4);
    _Float16* hA   = (_Float16*)alloc((size_t)NNODES * 64 * 2);
    _Float16* hB   = (_Float16*)alloc((size_t)NNODES * 64 * 2);
    float*    hw   = (float*)alloc((size_t)NNODES * 64 * 4);
    float*    acc  = (float*)alloc((size_t)NNODES * 64 * 4);

    const int NB = (NNODES + 255) / 256;
    const int GB = NTILES / 8;   // 1296 blocks, 8 waves/block, 1 tile/wave

    // degree -> dinv (shared by all layers)
    k_init_deg<<<NB, 256, 0, stream>>>(dinv);
    {
        dim3 g((EPER + 255) / 256, NPUZ);
        k_count_deg<<<g, 256, 0, stream>>>(ei, dinv);
    }
    k_rsqrt_inplace<<<NB, 256, 0, stream>>>(dinv);

    // x (f32, N x 10) -> hA (f16, N x 16 padded)
    k_cvt_x<<<(NNODES * 16 + 255) / 256, 256, 0, stream>>>(x, hA);

#define GCN_LAYER(CIN, KREAL, COUT, LOG2C, HIN, HOUT, Wp, bp)                          \
    k_gemm<CIN, KREAL, COUT, false, false, false><<<GB, 256, 0, stream>>>(             \
        HIN, Wp, nullptr, hw, nullptr, COUT);                                          \
    hipMemsetAsync(acc, 0, (size_t)NNODES * COUT * 4, stream);                         \
    {                                                                                  \
        dim3 g((EPER * COUT + 255) / 256, NPUZ);                                       \
        k_aggregate<LOG2C><<<g, 256, 0, stream>>>(ei, dinv, hw, acc);                  \
    }                                                                                  \
    k_finalize<LOG2C><<<(NNODES * COUT + 255) / 256, 256, 0, stream>>>(                \
        acc, hw, dinv, bp, HOUT);

    GCN_LAYER(16, 10, 16, 4, hA, hB, W1, b1)   // 10->16 (K padded to 16)
    GCN_LAYER(16, 16, 32, 5, hB, hA, W2, b2)   // 16->32
    GCN_LAYER(32, 32, 64, 6, hA, hB, W3, b3)   // 32->64
    GCN_LAYER(64, 64, 32, 5, hB, hA, W4, b4)   // 64->32
    GCN_LAYER(32, 32, 16, 4, hA, hB, W5, b5)   // 32->16
#undef GCN_LAYER

    // FC head: 16 -> 16 (relu) -> 9, bias folded into C fragment
    k_gemm<16, 16, 16, true, true, true><<<GB, 256, 0, stream>>>(
        hB, fW1, fb1, nullptr, hA, 16);
    k_gemm<16, 16, 9, false, false, true><<<GB, 256, 0, stream>>>(
        hA, fW2, fb2, out, nullptr, 9);
}
